// PicoMLA_81003083203598
// MI455X (gfx1250) — matrix operand order repaired and checked by reading, not compiled, not run on hardware
//
#include <hip/hip_runtime.h>
#include <stdint.h>

// ---------------- types ----------------
typedef __bf16 bf16_t;
typedef __attribute__((ext_vector_type(16))) __bf16 v16bf;
typedef __attribute__((ext_vector_type(8)))  float  v8f;

// ---------------- problem dims ----------------
#define BB   4
#define TT   2048
#define DD   2048
#define HH   16
#define NOPE 64
#define ROPE 64
#define VDIM 128
#define QKD  128
#define KV_R 512
#define Q_R  1024
#define EPSF 1e-6f
#define SCALEF 0.08838834764831845f   // 1/sqrt(128)

// ---------------- bf16 helpers ----------------
__device__ __forceinline__ float bf2f(bf16_t v) {
  unsigned short u = __builtin_bit_cast(unsigned short, v);
  unsigned int x = ((unsigned int)u) << 16;
  return __builtin_bit_cast(float, x);
}
__device__ __forceinline__ bf16_t f2bf(float f) {
  unsigned int x = __builtin_bit_cast(unsigned int, f);
  unsigned int r = x + 0x7FFFu + ((x >> 16) & 1u);   // round-to-nearest-even
  unsigned short u = (unsigned short)(r >> 16);
  return __builtin_bit_cast(bf16_t, u);
}

// ---------------- CDNA5 async global->LDS copy (ASYNCcnt path) ----------------
// Generic LDS pointer low 32 bits == workgroup LDS byte offset (ISA: LDS aperture
// addr = {SHARED_BASE, offset[31:0]}). Each lane moves 16 bytes; one instruction
// per wave moves 512B straight into LDS without touching VGPRs.
__device__ __forceinline__ void async_ld_b128(bf16_t* lds, const bf16_t* g) {
  unsigned int l = (unsigned int)(uintptr_t)lds;
  unsigned long long ga = (unsigned long long)(uintptr_t)g;
  asm volatile("global_load_async_to_lds_b128 %0, %1, off"
               :: "v"(l), "v"(ga) : "memory");
}
__device__ __forceinline__ void wait_async0() {
  asm volatile("s_wait_asynccnt 0" ::: "memory");
}

// ---------------- elementwise cast f32 -> bf16 ----------------
__global__ void mla_cast_bf16(const float* __restrict__ src, bf16_t* __restrict__ dst, long n) {
  long i = (long)blockIdx.x * blockDim.x + threadIdx.x;
  long stride = (long)gridDim.x * blockDim.x;
  for (; i < n; i += stride) dst[i] = f2bf(src[i]);
}

// cast + transpose: dst[n*K + k] = src[k*N + n]   (weights -> [N][K] bf16)
__global__ void mla_cast_bf16_tr(const float* __restrict__ src, bf16_t* __restrict__ dst,
                                 int K, int N) {
  const long total = (long)K * N;
  long i = (long)blockIdx.x * blockDim.x + threadIdx.x;
  const long stride = (long)gridDim.x * blockDim.x;
  for (; i < total; i += stride) {
    const long n = i / K;
    const long k = i - n * K;
    dst[i] = f2bf(src[k * (long)N + n]);
  }
}

// ---------------- WMMA GEMM: C(f32)[M,N] = A(bf16)[M,K] * Bt(bf16)[N,K] ----------------
// block tile 128x64, 8 waves; double-buffered LDS filled by async DMA.
#define GM_BM 128
#define GM_BN 64
#define GM_BK 32
#define AS_LD 40   // 32 + 8 pad (bf16), rows stay 16B aligned
#define BS_LD 40

__global__ __launch_bounds__(256)
void mla_gemm_bf16(const bf16_t* __restrict__ A, const bf16_t* __restrict__ Bt,
                   float* __restrict__ C, int M, int N, int K) {
  __shared__ bf16_t As[2][GM_BM * AS_LD];
  __shared__ bf16_t Bs[2][GM_BN * BS_LD];   // [n][k]

  const int t    = threadIdx.x;
  const int wave = t >> 5;
  const int lane = t & 31;
  const int mbase = blockIdx.y * GM_BM;
  const int nbase = blockIdx.x * GM_BN;

  v8f acc[4];
  #pragma unroll
  for (int nt = 0; nt < 4; ++nt)
    #pragma unroll
    for (int r = 0; r < 8; ++r) acc[nt][r] = 0.0f;

  // async staging map: 16B chunks, 4 per 32-elem row
  const int sr = t >> 2;          // 0..63
  const int sg = (t & 3) * 8;     // elem offset 0,8,16,24

  const int rowA = lane & 15;
  const int kbA  = (lane < 16) ? 0 : 16;
  const int colB = lane & 15;
  const int kbB  = (lane < 16) ? 0 : 16;

  auto stage = [&](int buf, int kk) {
    async_ld_b128(&As[buf][sr * AS_LD + sg],
                  A + (size_t)(mbase + sr) * K + kk + sg);
    async_ld_b128(&As[buf][(sr + 64) * AS_LD + sg],
                  A + (size_t)(mbase + sr + 64) * K + kk + sg);
    async_ld_b128(&Bs[buf][sr * BS_LD + sg],
                  Bt + (size_t)(nbase + sr) * K + kk + sg);
  };

  stage(0, 0);
  wait_async0();
  __syncthreads();

  const int niter = K / GM_BK;
  for (int i = 0; i < niter; ++i) {
    const int cur = i & 1;
    if (i + 1 < niter) stage(cur ^ 1, (i + 1) * GM_BK);   // prefetch next tile via DMA

    // A fragment for this wave's 16 rows
    v16bf afr;
    {
      const bf16_t* p = &As[cur][(wave * 16 + rowA) * AS_LD];
      #pragma unroll
      for (int e = 0; e < 8; ++e) afr[e] = p[kbA + e];
      #pragma unroll
      for (int e = 0; e < 8; ++e) afr[8 + e] = p[kbA + 8 + e];
    }
    #pragma unroll
    for (int nt = 0; nt < 4; ++nt) {
      v16bf bfr;
      const bf16_t* q = &Bs[cur][(nt * 16 + colB) * BS_LD + kbB];
      #pragma unroll
      for (int e = 0; e < 16; ++e) bfr[e] = q[e];
      acc[nt] = __builtin_amdgcn_wmma_f32_16x16x32_bf16(
          false, afr, false, bfr, (short)0, acc[nt], false, false);
    }
    wait_async0();
    __syncthreads();
  }

  const int colC = lane & 15;
  const int rbC  = (lane < 16) ? 0 : 8;
  #pragma unroll
  for (int nt = 0; nt < 4; ++nt)
    #pragma unroll
    for (int r = 0; r < 8; ++r)
      C[(size_t)(mbase + wave * 16 + rbC + r) * N + nbase + nt * 16 + colC] = acc[nt][r];
}

// ---------------- row RMSNorm (f32 in -> bf16 out) ----------------
__global__ __launch_bounds__(256)
void mla_rmsnorm_rows(const float* __restrict__ in, const float* __restrict__ g,
                      bf16_t* __restrict__ out, int N) {
  const int row = blockIdx.x;
  const float* x = in + (size_t)row * N;
  __shared__ float red[256];
  float s = 0.0f;
  for (int i = threadIdx.x; i < N; i += 256) { float v = x[i]; s += v * v; }
  red[threadIdx.x] = s; __syncthreads();
  for (int o = 128; o > 0; o >>= 1) {
    if (threadIdx.x < o) red[threadIdx.x] += red[threadIdx.x + o];
    __syncthreads();
  }
  const float inv = rsqrtf(red[0] / (float)N + EPSF);
  for (int i = threadIdx.x; i < N; i += 256) out[(size_t)row * N + i] = f2bf(x[i] * inv * g[i]);
}

// ---------------- build q: RoPE + per-head RMSNorm, layout [B,H,T,128] ----------------
__global__ __launch_bounds__(128)
void mla_build_q(const float* __restrict__ q_raw, const float* __restrict__ g_q,
                 bf16_t* __restrict__ q_bf) {
  const int idx  = blockIdx.x;          // bt*H + h
  const int h    = idx % HH;
  const int bt   = idx / HH;
  const int tpos = bt % TT;
  const int b    = bt / TT;
  const int d    = threadIdx.x;
  const float* src = q_raw + (size_t)bt * (HH * QKD) + h * QKD;
  float val;
  if (d < NOPE) {
    val = src[d];
  } else {
    const int j = d - NOPE;
    const float invf = __powf(10000.0f, -(float)(j & 31) / 32.0f);
    const float ang = (float)tpos * invf;
    const float c = __cosf(ang), s = __sinf(ang);
    const float x = src[NOPE + j];
    const float other = src[NOPE + ((j < 32) ? j + 32 : j - 32)];
    const float rot = (j < 32) ? -other : other;
    val = x * c + rot * s;
  }
  __shared__ float red[128];
  red[d] = val * val; __syncthreads();
  for (int o = 64; o > 0; o >>= 1) {
    if (d < o) red[d] += red[d + o];
    __syncthreads();
  }
  const float inv = rsqrtf(red[0] / (float)QKD + EPSF);
  q_bf[(((size_t)b * HH + h) * TT + tpos) * QKD + d] = f2bf(val * inv * g_q[d]);
}

// ---------------- build k: concat(k_nope, rope(k_rope)) + RMSNorm, [B,H,S,128] ----------------
__global__ __launch_bounds__(128)
void mla_build_k(const float* __restrict__ k_nope_raw, const float* __restrict__ k_rope_raw,
                 const float* __restrict__ g_k, bf16_t* __restrict__ k_bf) {
  const int idx  = blockIdx.x;
  const int h    = idx % HH;
  const int bt   = idx / HH;
  const int spos = bt % TT;
  const int b    = bt / TT;
  const int d    = threadIdx.x;
  float val;
  if (d < NOPE) {
    val = k_nope_raw[(size_t)bt * (HH * NOPE) + h * NOPE + d];
  } else {
    const int j = d - NOPE;
    const float invf = __powf(10000.0f, -(float)(j & 31) / 32.0f);
    const float ang = (float)spos * invf;
    const float c = __cosf(ang), s = __sinf(ang);
    const float x = k_rope_raw[(size_t)bt * ROPE + j];
    const float other = k_rope_raw[(size_t)bt * ROPE + ((j < 32) ? j + 32 : j - 32)];
    const float rot = (j < 32) ? -other : other;
    val = x * c + rot * s;
  }
  __shared__ float red[128];
  red[d] = val * val; __syncthreads();
  for (int o = 64; o > 0; o >>= 1) {
    if (d < o) red[d] += red[d + o];
    __syncthreads();
  }
  const float inv = rsqrtf(red[0] / (float)QKD + EPSF);
  k_bf[(((size_t)b * HH + h) * TT + spos) * QKD + d] = f2bf(val * inv * g_k[d]);
}

// ---------------- v: [bt, h*128+d] -> bf16 transposed per head [B,H,d,S] ----------------
__global__ void mla_build_vtr(const float* __restrict__ v_raw, bf16_t* __restrict__ v_tr) {
  const long total = (long)BB * TT * HH * VDIM;
  long i = (long)blockIdx.x * blockDim.x + threadIdx.x;
  const long stride = (long)gridDim.x * blockDim.x;
  for (; i < total; i += stride) {
    const int d = (int)(i % VDIM);
    long r = i / VDIM;
    const int h = (int)(r % HH);
    const long bt = r / HH;
    const int tpos = (int)(bt % TT);
    const int b = (int)(bt / TT);
    v_tr[(((size_t)b * HH + h) * VDIM + d) * TT + tpos] = f2bf(v_raw[i]);
  }
}

// ---------------- causal flash attention, async-staged K/V tiles ----------------
#define KS_LD 136   // 128 + 8 pad, rows 16B aligned
#define VT_LD 40    // 32 + 8 pad

__global__ __launch_bounds__(256)
void mla_flash(const bf16_t* __restrict__ q_bf, const bf16_t* __restrict__ k_bf,
               const bf16_t* __restrict__ v_tr, float* __restrict__ attn_out) {
  __shared__ bf16_t Ks[32 * KS_LD];      // [s_local][kdim]
  __shared__ bf16_t Vt[128 * VT_LD];     // [vdim][s_local]
  __shared__ bf16_t Plds[8 * 16 * 32];   // per-wave P tile, [m][s_local]

  const int t    = threadIdx.x;
  const int wave = t >> 5;
  const int lane = t & 31;
  const int qb   = blockIdx.x * 128;
  const int h    = blockIdx.y;
  const int b    = blockIdx.z;
  const size_t base    = ((size_t)b * HH + h) * (size_t)TT * QKD;
  const size_t vtrbase = ((size_t)b * HH + h) * (size_t)VDIM * TT;

  const int row = lane & 15;
  const int kbA = (lane < 16) ? 0 : 16;
  const int col = lane & 15;
  const int rb  = (lane < 16) ? 0 : 8;
  const int kbB = (lane < 16) ? 0 : 16;

  // preload Q fragments
  v16bf qf[4];
  {
    const bf16_t* qp = q_bf + base + (size_t)(qb + wave * 16 + row) * QKD;
    #pragma unroll
    for (int c = 0; c < 4; ++c) {
      #pragma unroll
      for (int e = 0; e < 8; ++e) qf[c][e] = qp[c * 32 + kbA + e];
      #pragma unroll
      for (int e = 0; e < 8; ++e) qf[c][8 + e] = qp[c * 32 + kbA + 8 + e];
    }
  }

  v8f o[8];
  float m_i[8], l_i[8];
  #pragma unroll
  for (int nt = 0; nt < 8; ++nt)
    #pragma unroll
    for (int r = 0; r < 8; ++r) o[nt][r] = 0.0f;
  #pragma unroll
  for (int r = 0; r < 8; ++r) { m_i[r] = -1e30f; l_i[r] = 0.0f; }

  const int s_end   = qb + 128;
  const int wave_hi = qb + wave * 16 + 15;
  const int kr = t >> 4, kc = (t & 15) * 8;  // K tile chunks
  const int vr = t >> 2, vc = (t & 3) * 8;   // V tile chunks

  for (int sb = 0; sb < s_end; sb += 32) {
    // async DMA staging of K (natural) and V (pre-transposed) tiles
    async_ld_b128(&Ks[kr * KS_LD + kc],        k_bf + base + (size_t)(sb + kr) * QKD + kc);
    async_ld_b128(&Ks[(kr + 16) * KS_LD + kc], k_bf + base + (size_t)(sb + kr + 16) * QKD + kc);
    async_ld_b128(&Vt[vr * VT_LD + vc],        v_tr + vtrbase + (size_t)vr * TT + sb + vc);
    async_ld_b128(&Vt[(vr + 64) * VT_LD + vc], v_tr + vtrbase + (size_t)(vr + 64) * TT + sb + vc);
    wait_async0();
    __syncthreads();

    if (sb <= wave_hi) {
      bf16_t* Pw = &Plds[wave * 16 * 32];
      #pragma unroll
      for (int half = 0; half < 2; ++half) {
        v8f sc;
        #pragma unroll
        for (int r = 0; r < 8; ++r) sc[r] = 0.0f;
        #pragma unroll
        for (int c = 0; c < 4; ++c) {
          v16bf bk;
          const bf16_t* kp = &Ks[(half * 16 + col) * KS_LD + c * 32 + kbB];
          #pragma unroll
          for (int e = 0; e < 16; ++e) bk[e] = kp[e];
          sc = __builtin_amdgcn_wmma_f32_16x16x32_bf16(
              false, qf[c], false, bk, (short)0, sc, false, false);
        }
        const int scol = sb + half * 16 + col;
        #pragma unroll
        for (int r = 0; r < 8; ++r) {
          const int qr = qb + wave * 16 + rb + r;
          float v = sc[r] * SCALEF;
          if (scol > qr) v = -1e30f;
          sc[r] = v;
        }
        #pragma unroll
        for (int r = 0; r < 8; ++r) {
          float rm = sc[r];
          #pragma unroll
          for (int off = 1; off < 16; off <<= 1) rm = fmaxf(rm, __shfl_xor(rm, off, 32));
          const float mn = fmaxf(m_i[r], rm);
          const float corr = __expf(m_i[r] - mn);
          const float p = __expf(sc[r] - mn);
          float ps = p;
          #pragma unroll
          for (int off = 1; off < 16; off <<= 1) ps += __shfl_xor(ps, off, 32);
          l_i[r] = l_i[r] * corr + ps;
          m_i[r] = mn;
          #pragma unroll
          for (int nt = 0; nt < 8; ++nt) o[nt][r] *= corr;
          Pw[(rb + r) * 32 + half * 16 + col] = f2bf(p);
        }
      }
      __builtin_amdgcn_sched_barrier(0);   // wave-local LDS RAW ordering

      v16bf aP;
      {
        const bf16_t* pp = &Pw[row * 32];
        #pragma unroll
        for (int e = 0; e < 8; ++e) aP[e] = pp[kbA + e];
        #pragma unroll
        for (int e = 0; e < 8; ++e) aP[8 + e] = pp[kbA + 8 + e];
      }
      #pragma unroll
      for (int nt = 0; nt < 8; ++nt) {
        v16bf bv;
        const bf16_t* vp = &Vt[(nt * 16 + col) * VT_LD + kbB];
        #pragma unroll
        for (int e = 0; e < 16; ++e) bv[e] = vp[e];
        o[nt] = __builtin_amdgcn_wmma_f32_16x16x32_bf16(
            false, aP, false, bv, (short)0, o[nt], false, false);
      }
    }
    __syncthreads();
  }

  float* outp = attn_out + ((size_t)b * HH + h) * (size_t)TT * VDIM;
  #pragma unroll
  for (int r = 0; r < 8; ++r) {
    const float invl = 1.0f / l_i[r];
    const int qr = qb + wave * 16 + rb + r;
    #pragma unroll
    for (int nt = 0; nt < 8; ++nt)
      outp[(size_t)qr * VDIM + nt * 16 + col] = o[nt][r] * invl;
  }
}

// ---------------- gate: y = attn * silu(gate), back to [bt, h*128+d], bf16 ----------------
__global__ void mla_gate_combine(const float* __restrict__ attn, const float* __restrict__ gate_raw,
                                 bf16_t* __restrict__ y_bf) {
  const long total = (long)BB * TT * HH * VDIM;
  long i = (long)blockIdx.x * blockDim.x + threadIdx.x;
  const long stride = (long)gridDim.x * blockDim.x;
  for (; i < total; i += stride) {
    const int d = (int)(i % VDIM);
    long r = i / VDIM;
    const int h = (int)(r % HH);
    const long bt = r / HH;
    const int tpos = (int)(bt % TT);
    const int b = (int)(bt / TT);
    const float a = attn[(((size_t)b * HH + h) * TT + tpos) * VDIM + d];
    const float g = gate_raw[i];
    const float sig = 1.0f / (1.0f + __expf(-g));
    y_bf[i] = f2bf(a * g * sig);
  }
}

// ---------------- host-side orchestration ----------------
extern "C" void kernel_launch(void* const* d_in, const int* in_sizes, int n_in,
                              void* d_out, int out_size, void* d_ws, size_t ws_size,
                              hipStream_t stream) {
  (void)in_sizes; (void)n_in; (void)out_size; (void)ws_size;

  const float* x         = (const float*)d_in[0];
  const float* w_q_down  = (const float*)d_in[1];
  const float* g_q_down  = (const float*)d_in[2];
  const float* w_q_up    = (const float*)d_in[3];
  const float* w_kv_down = (const float*)d_in[4];
  const float* g_kv_down = (const float*)d_in[5];
  const float* w_k_up    = (const float*)d_in[6];
  const float* w_v_up    = (const float*)d_in[7];
  const float* w_k_rope  = (const float*)d_in[8];
  const float* g_q       = (const float*)d_in[9];
  const float* g_k       = (const float*)d_in[10];
  const float* w_gate    = (const float*)d_in[11];
  const float* w_o       = (const float*)d_in[12];
  float* out = (float*)d_out;

  const long M = (long)BB * TT;   // 8192 token rows

  char* ws = (char*)d_ws;
  size_t off = 0;
  auto alloc = [&](size_t bytes) -> void* {
    void* p = ws + off;
    off = (off + bytes + 255) & ~(size_t)255;
    return p;
  };

  bf16_t* xb     = (bf16_t*)alloc((size_t)M * DD * 2);
  // transposed bf16 weights: [N][K]
  bf16_t* wqd_t  = (bf16_t*)alloc((size_t)Q_R * DD * 2);
  bf16_t* wqu_t  = (bf16_t*)alloc((size_t)HH * QKD * Q_R * 2);
  bf16_t* wkvd_t = (bf16_t*)alloc((size_t)KV_R * DD * 2);
  bf16_t* wku_t  = (bf16_t*)alloc((size_t)HH * NOPE * KV_R * 2);
  bf16_t* wvu_t  = (bf16_t*)alloc((size_t)HH * VDIM * KV_R * 2);
  bf16_t* wkr_t  = (bf16_t*)alloc((size_t)ROPE * DD * 2);
  bf16_t* wg_t   = (bf16_t*)alloc((size_t)HH * VDIM * DD * 2);
  bf16_t* wo_t   = (bf16_t*)alloc((size_t)DD * HH * VDIM * 2);

  float* q_c_raw    = (float*)alloc((size_t)M * Q_R * 4);
  float* c_kv_raw   = (float*)alloc((size_t)M * KV_R * 4);
  float* k_rope_raw = (float*)alloc((size_t)M * ROPE * 4);
  float* gate_raw   = (float*)alloc((size_t)M * HH * VDIM * 4);
  bf16_t* q_cb      = (bf16_t*)alloc((size_t)M * Q_R * 2);
  bf16_t* c_kvb     = (bf16_t*)alloc((size_t)M * KV_R * 2);
  float* q_raw      = (float*)alloc((size_t)M * HH * QKD * 4);
  float* k_nope_raw = (float*)alloc((size_t)M * HH * NOPE * 4);
  float* v_raw      = (float*)alloc((size_t)M * HH * VDIM * 4);
  bf16_t* q_bf      = (bf16_t*)alloc((size_t)M * HH * QKD * 2);
  bf16_t* k_bf      = (bf16_t*)alloc((size_t)M * HH * QKD * 2);
  bf16_t* v_tr      = (bf16_t*)alloc((size_t)M * HH * VDIM * 2);
  float* attn       = (float*)alloc((size_t)M * HH * VDIM * 4);
  bf16_t* y_bf      = (bf16_t*)alloc((size_t)M * HH * VDIM * 2);

  // 1. cast x; cast+transpose weights
  mla_cast_bf16<<<4096, 256, 0, stream>>>(x, xb, M * (long)DD);
  auto castT = [&](const float* s, bf16_t* d, int K, int N) {
    mla_cast_bf16_tr<<<4096, 256, 0, stream>>>(s, d, K, N);
  };
  castT(w_q_down,  wqd_t,  DD,   Q_R);
  castT(w_q_up,    wqu_t,  Q_R,  HH * QKD);
  castT(w_kv_down, wkvd_t, DD,   KV_R);
  castT(w_k_up,    wku_t,  KV_R, HH * NOPE);
  castT(w_v_up,    wvu_t,  KV_R, HH * VDIM);
  castT(w_k_rope,  wkr_t,  DD,   ROPE);
  castT(w_gate,    wg_t,   DD,   HH * VDIM);
  castT(w_o,       wo_t,   HH * VDIM, DD);

  auto gemm = [&](const bf16_t* A, const bf16_t* Bt, float* C, int m, int n, int k) {
    dim3 grid(n / GM_BN, m / GM_BM);
    mla_gemm_bf16<<<grid, 256, 0, stream>>>(A, Bt, C, m, n, k);
  };

  // 2. down-projections + rope-k + gate from x
  gemm(xb, wqd_t,  q_c_raw,    (int)M, Q_R,       DD);
  gemm(xb, wkvd_t, c_kv_raw,   (int)M, KV_R,      DD);
  gemm(xb, wkr_t,  k_rope_raw, (int)M, ROPE,      DD);
  gemm(xb, wg_t,   gate_raw,   (int)M, HH * VDIM, DD);

  // 3. RMSNorm on latent vectors
  mla_rmsnorm_rows<<<(int)M, 256, 0, stream>>>(q_c_raw,  g_q_down,  q_cb,  Q_R);
  mla_rmsnorm_rows<<<(int)M, 256, 0, stream>>>(c_kv_raw, g_kv_down, c_kvb, KV_R);

  // 4. up-projections
  gemm(q_cb,  wqu_t, q_raw,      (int)M, HH * QKD,  Q_R);
  gemm(c_kvb, wku_t, k_nope_raw, (int)M, HH * NOPE, KV_R);
  gemm(c_kvb, wvu_t, v_raw,      (int)M, HH * VDIM, KV_R);

  // 5. RoPE + per-head RMSNorm + head-major relayout
  mla_build_q<<<(int)(M * HH), 128, 0, stream>>>(q_raw, g_q, q_bf);
  mla_build_k<<<(int)(M * HH), 128, 0, stream>>>(k_nope_raw, k_rope_raw, g_k, k_bf);
  mla_build_vtr<<<4096, 256, 0, stream>>>(v_raw, v_tr);

  // 6. causal flash attention (WMMA QK^T and P.V, async K/V staging)
  {
    dim3 grid(TT / 128, HH, BB);
    mla_flash<<<grid, 256, 0, stream>>>(q_bf, k_bf, v_tr, attn);
  }

  // 7. SiLU gate + relayout
  mla_gate_combine<<<4096, 256, 0, stream>>>(attn, gate_raw, y_bf);

  // 8. output projection -> d_out (f32)
  gemm(y_bf, wo_t, out, (int)M, DD, HH * VDIM);
}
